// SparseAutoencoder_41798621725055
// MI455X (gfx1250) — matrix operand order repaired and checked by reading, not compiled, not run on hardware
//
#include <hip/hip_runtime.h>
#include <hip/hip_bf16.h>

// ---------------- problem dims ----------------
#define NROWS  8192
#define HID    1024
#define SP     32768
#define KMAX   64
#define LN_EPS 1e-5f

// ---------------- WMMA vector types ----------------
typedef __attribute__((ext_vector_type(16))) __bf16 v16bf;
typedef __attribute__((ext_vector_type(8)))  float  v8f;

__device__ __forceinline__ unsigned short f32_to_bf16(float f) {
    unsigned int u = __float_as_uint(f);
    unsigned int r = (u + 0x7FFFu + ((u >> 16) & 1u)) >> 16;   // round-to-nearest-even
    return (unsigned short)r;
}

// ============================================================
// 1) LayerNorm (fp32) -> h in bf16   [NROWS x HID]
// ============================================================
__global__ __launch_bounds__(256) void ln_bf16_kernel(
    const float* __restrict__ x, const float* __restrict__ gamma,
    const float* __restrict__ beta, unsigned short* __restrict__ hB)
{
    __shared__ float red[256];
    const int row = blockIdx.x;
    const int tid = threadIdx.x;
    const float* xr = x + (size_t)row * HID;

    float4 xv = *(const float4*)(xr + tid * 4);
    red[tid] = xv.x + xv.y + xv.z + xv.w; __syncthreads();
    for (int off = 128; off > 0; off >>= 1) {
        if (tid < off) red[tid] += red[tid + off];
        __syncthreads();
    }
    const float mean = red[0] * (1.0f / HID);
    __syncthreads();

    float d0 = xv.x - mean, d1 = xv.y - mean, d2 = xv.z - mean, d3 = xv.w - mean;
    red[tid] = d0*d0 + d1*d1 + d2*d2 + d3*d3; __syncthreads();
    for (int off = 128; off > 0; off >>= 1) {
        if (tid < off) red[tid] += red[tid + off];
        __syncthreads();
    }
    const float rstd = rsqrtf(red[0] * (1.0f / HID) + LN_EPS);

    float4 gv = *(const float4*)(gamma + tid * 4);
    float4 bv = *(const float4*)(beta  + tid * 4);
    ushort4 o;
    o.x = f32_to_bf16(d0 * rstd * gv.x + bv.x);
    o.y = f32_to_bf16(d1 * rstd * gv.y + bv.y);
    o.z = f32_to_bf16(d2 * rstd * gv.z + bv.z);
    o.w = f32_to_bf16(d3 * rstd * gv.w + bv.w);
    *(ushort4*)(hB + (size_t)row * HID + tid * 4) = o;
}

// ============================================================
// 2) fp32 -> bf16 cast of W_enc  (64 MB result: L2-resident)
// ============================================================
__global__ void cast_bf16_kernel(const float* __restrict__ in,
                                 unsigned short* __restrict__ out, size_t n)
{
    size_t i = (size_t)blockIdx.x * blockDim.x + threadIdx.x;
    const size_t stride = (size_t)gridDim.x * blockDim.x;
    for (; i < n; i += stride) out[i] = f32_to_bf16(in[i]);
}

// ============================================================
// 3) W_dec [HID x SP] -> W_decT [SP x HID]  (tile transpose)
// ============================================================
__global__ __launch_bounds__(256) void transpose_kernel(
    const float* __restrict__ in, float* __restrict__ out)
{
    __shared__ float tile[32][33];
    const int x  = blockIdx.x * 32 + threadIdx.x;   // sparse index
    const int y0 = blockIdx.y * 32;                 // hidden index
    for (int r = threadIdx.y; r < 32; r += 8)
        tile[r][threadIdx.x] = in[(size_t)(y0 + r) * SP + x];
    __syncthreads();
    const int xo  = blockIdx.y * 32 + threadIdx.x;  // hidden
    const int yo0 = blockIdx.x * 32;
    for (int r = threadIdx.y; r < 32; r += 8)
        out[(size_t)(yo0 + r) * HID + xo] = tile[threadIdx.x][r];
}

// ============================================================
// 4) Encode GEMM:  act = relu(h @ W_enc^T + b_enc)   [NROWS x SP]
//    bf16 WMMA 16x16x32. Block = 8 waves = 128M x 128N tile.
//    B tile (128x32 bf16 = 8 KB/K-step) double-buffered in LDS:
//    next K-slice global->regs overlaps current-slice WMMAs,
//    B fragments come from ds_load_b128 (shared by all 8 waves).
// ============================================================
__global__ __launch_bounds__(256) void encode_wmma_kernel(
    const unsigned short* __restrict__ hB,
    const unsigned short* __restrict__ WencB,
    const float* __restrict__ b_enc,
    float* __restrict__ act)
{
    __shared__ __align__(16) unsigned short bsm[2][128 * 32];   // 2 x 8 KB

    const int tid  = threadIdx.x;
    const int wave = tid >> 5;
    const int lane = tid & 31;
    const int lm   = lane & 15;     // M (or N) within 16x16 tile
    const int half = lane >> 4;     // K-half selector per ISA layout

    const int m_base = blockIdx.y * 128 + wave * 16;
    const int n_base = blockIdx.x * 128;

    // cooperative B staging: 2 threads per row, 32 B (16 ushorts) each
    const int srow  = tid >> 1;     // 0..127
    const int spart = tid & 1;      // 0..1
    const unsigned short* gB =
        WencB + (size_t)(n_base + srow) * HID + spart * 16;
    const int soff = srow * 32 + spart * 16;

    v8f acc[8] = {};   // 8 accumulators: 16M x 128N per wave
    const unsigned short* arow = hB + (size_t)(m_base + lm) * HID;

    // prologue: stage K-step 0
    {
        uint4 s0 = *(const uint4*)(gB + 0);
        uint4 s1 = *(const uint4*)(gB + 8);
        *(uint4*)(&bsm[0][soff + 0]) = s0;
        *(uint4*)(&bsm[0][soff + 8]) = s1;
    }
    __syncthreads();

    for (int ks = 0; ks < 32; ++ks) {
        const int k0  = ks * 32;
        const int cur = ks & 1;

        // kick off global loads of the NEXT B slice while WMMAs run
        uint4 n0, n1;
        if (ks < 31) {
            n0 = *(const uint4*)(gB + k0 + 32);
            n1 = *(const uint4*)(gB + k0 + 40);
            __builtin_prefetch(arow + k0 + 64, 0, 0);   // global_prefetch_b8
        }

        // A fragment 16x32 bf16 (direct global, zero redundancy):
        //   lanes 0-15 : K=k0+{0..15}; lanes 16-31: K=k0+16+{0..15}
        v16bf a;
        ((uint4*)&a)[0] = *(const uint4*)(arow + k0 + half * 16);
        ((uint4*)&a)[1] = *(const uint4*)(arow + k0 + half * 16 + 8);

        #pragma unroll
        for (int j = 0; j < 8; ++j) {
            // B fragment 32x16: lane = column N=lm, K = k0 + half*16 + {0..15}
            const unsigned short* bls = &bsm[cur][(j * 16 + lm) * 32 + half * 16];
            v16bf b;
            ((uint4*)&b)[0] = *(const uint4*)(bls);
            ((uint4*)&b)[1] = *(const uint4*)(bls + 8);
            acc[j] = __builtin_amdgcn_wmma_f32_16x16x32_bf16(
                         false, a, false, b, (short)0, acc[j], false, false);
        }

        // commit next slice into the other buffer
        if (ks < 31) {
            *(uint4*)(&bsm[cur ^ 1][soff + 0]) = n0;
            *(uint4*)(&bsm[cur ^ 1][soff + 8]) = n1;
        }
        __syncthreads();   // single barrier/step is safe with 2 buffers
    }

    // Epilogue: bias + ReLU, scatter per ISA C/D layout
    //   VGPR i, lanes 0-15: (M=i, N=lane); lanes 16-31: (M=i+8, N=lane-16)
    #pragma unroll
    for (int j = 0; j < 8; ++j) {
        const int col  = n_base + j * 16 + lm;
        const float bb = b_enc[col];
        #pragma unroll
        for (int i = 0; i < 8; ++i) {
            const int row = m_base + i + half * 8;
            float v = acc[j][i] + bb;
            act[(size_t)row * SP + col] = v > 0.0f ? v : 0.0f;
        }
    }
}

// ============================================================
// 5) Top-k radix select per row, in place on codes (d_out).
//    Row (128 KB) cached in LDS (320 KB/WGP available).
//    Values are >= 0 post-ReLU -> uint bit order == float order.
// ============================================================
__global__ __launch_bounds__(256) void topk_kernel(
    float* __restrict__ codes, float* __restrict__ idx_out,
    const int* __restrict__ kptr)
{
    extern __shared__ unsigned int sm[];
    unsigned int* v      = sm;              // SP values (bit patterns)
    unsigned int* hist   = v + SP;          // 256 bins
    unsigned int* selIdx = hist + 256;      // 64
    unsigned int* selVal = selIdx + 64;     // 64
    unsigned int* ctl    = selVal + 64;     // [0]=prefix [1]=kneed [2]=slot [3]=eq

    const int tid = threadIdx.x;
    const int row = blockIdx.x;
    int kk = *kptr; if (kk > KMAX) kk = KMAX; if (kk < 1) kk = 1;

    float* rowp = codes + (size_t)row * SP;
    for (int i = tid; i < SP; i += 256) v[i] = __float_as_uint(rowp[i]);
    if (tid == 0) { ctl[0] = 0; ctl[1] = (unsigned)kk; ctl[2] = 0; ctl[3] = 0; }
    __syncthreads();

    unsigned int pmask = 0;
    for (int shift = 24; shift >= 0; shift -= 8) {
        hist[tid] = 0;
        __syncthreads();
        const unsigned int prefix = ctl[0];
        const unsigned int kneed  = ctl[1];
        for (int i = tid; i < SP; i += 256) {
            const unsigned int b = v[i];
            if ((b & pmask) == prefix) atomicAdd(&hist[(b >> shift) & 255u], 1u);
        }
        __syncthreads();
        if (tid == 0) {
            unsigned int cum = 0; int bin = 0;
            for (int b2 = 255; b2 >= 0; --b2) {
                const unsigned int c = hist[b2];
                if (cum + c >= kneed) { bin = b2; ctl[1] = kneed - cum; break; }
                cum += c;
            }
            ctl[0] = prefix | ((unsigned)bin << shift);
        }
        __syncthreads();
        pmask |= (255u << shift);
    }

    const unsigned int T       = ctl[0];   // exact k-th value (bits)
    const unsigned int need_eq = ctl[1];   // how many ==T survive

    for (int i = tid; i < SP; i += 256) {
        const unsigned int b = v[i];
        int slot = -1;
        if (b > T) {
            slot = (int)atomicAdd(&ctl[2], 1u);
        } else if (b == T) {
            const unsigned int e = atomicAdd(&ctl[3], 1u);
            if (e < need_eq) slot = (int)atomicAdd(&ctl[2], 1u);
        }
        if (slot >= 0 && slot < KMAX) { selIdx[slot] = (unsigned)i; selVal[slot] = b; }
        rowp[i] = (slot >= 0) ? __uint_as_float(b) : 0.0f;   // scatter codes in place
    }
    __syncthreads();

    // order like top_k: descending value, ties -> lower index first
    if (tid < kk) {
        const unsigned int myv = selVal[tid], myi = selIdx[tid];
        int rank = 0;
        for (int j = 0; j < kk; ++j) {
            const unsigned int vj = selVal[j], ij = selIdx[j];
            rank += (vj > myv) || (vj == myv && ij < myi);
        }
        idx_out[(size_t)row * kk + rank] = (float)myi;
    }
}

// ============================================================
// 6) Sparse decode: rec[row,:] = sum_j val_j * W_dec[:, idx_j]
// ============================================================
__global__ __launch_bounds__(256) void decode_t_kernel(
    const float* __restrict__ codes, const float* __restrict__ idxf,
    const float* __restrict__ WdecT, float* __restrict__ rec,
    const int* __restrict__ kptr)
{
    __shared__ int   s_idx[KMAX];
    __shared__ float s_val[KMAX];
    const int tid = threadIdx.x;
    const int row = blockIdx.x;
    int kk = *kptr; if (kk > KMAX) kk = KMAX; if (kk < 1) kk = 1;

    if (tid < kk) {
        const int s = (int)idxf[(size_t)row * kk + tid];
        s_idx[tid] = s;
        s_val[tid] = codes[(size_t)row * SP + s];
    }
    __syncthreads();

    float a0 = 0.f, a1 = 0.f, a2 = 0.f, a3 = 0.f;
    for (int j = 0; j < kk; ++j) {
        const float  c = s_val[j];
        const float4 w = *(const float4*)(WdecT + (size_t)s_idx[j] * HID + tid * 4);
        a0 += c * w.x; a1 += c * w.y; a2 += c * w.z; a3 += c * w.w;
    }
    float4 o; o.x = a0; o.y = a1; o.z = a2; o.w = a3;
    *(float4*)(rec + (size_t)row * HID + tid * 4) = o;
}

__global__ __launch_bounds__(256) void decode_nt_kernel(
    const float* __restrict__ codes, const float* __restrict__ idxf,
    const float* __restrict__ Wdec, float* __restrict__ rec,
    const int* __restrict__ kptr)
{
    __shared__ int   s_idx[KMAX];
    __shared__ float s_val[KMAX];
    const int tid = threadIdx.x;
    const int row = blockIdx.x;
    int kk = *kptr; if (kk > KMAX) kk = KMAX; if (kk < 1) kk = 1;

    if (tid < kk) {
        const int s = (int)idxf[(size_t)row * kk + tid];
        s_idx[tid] = s;
        s_val[tid] = codes[(size_t)row * SP + s];
    }
    __syncthreads();

    float a0 = 0.f, a1 = 0.f, a2 = 0.f, a3 = 0.f;
    const int h0 = tid * 4;
    for (int j = 0; j < kk; ++j) {
        const float c = s_val[j];
        const size_t s = (size_t)s_idx[j];
        a0 += c * Wdec[(size_t)(h0 + 0) * SP + s];
        a1 += c * Wdec[(size_t)(h0 + 1) * SP + s];
        a2 += c * Wdec[(size_t)(h0 + 2) * SP + s];
        a3 += c * Wdec[(size_t)(h0 + 3) * SP + s];
    }
    float4 o; o.x = a0; o.y = a1; o.z = a2; o.w = a3;
    *(float4*)(rec + (size_t)row * HID + h0) = o;
}

// ============================================================
// launcher
// ============================================================
extern "C" void kernel_launch(void* const* d_in, const int* in_sizes, int n_in,
                              void* d_out, int out_size, void* d_ws, size_t ws_size,
                              hipStream_t stream)
{
    const float* x     = (const float*)d_in[0];
    const float* gamma = (const float*)d_in[1];
    const float* beta  = (const float*)d_in[2];
    const float* W_enc = (const float*)d_in[3];
    const float* b_enc = (const float*)d_in[4];
    const float* W_dec = (const float*)d_in[5];
    const int*   kptr  = (const int*)d_in[6];

    // outputs: reconstruction [N,H] | sparse_codes [N,S] | idx [N,k] (as float)
    float* rec   = (float*)d_out;
    float* codes = rec + (size_t)NROWS * HID;
    float* idxf  = codes + (size_t)NROWS * SP;

    // workspace: h(bf16) | W_enc(bf16) | W_decT(f32)
    unsigned short* hB    = (unsigned short*)d_ws;
    unsigned short* WencB = hB + (size_t)NROWS * HID;
    float*          WdecT = (float*)(WencB + (size_t)SP * HID);
    const size_t need_tr = (size_t)NROWS * HID * 2 + (size_t)SP * HID * 2
                         + (size_t)SP * HID * 4;
    const bool use_tr = (ws_size >= need_tr);

    ln_bf16_kernel<<<NROWS, 256, 0, stream>>>(x, gamma, beta, hB);
    cast_bf16_kernel<<<4096, 256, 0, stream>>>(W_enc, WencB, (size_t)SP * HID);
    if (use_tr)
        transpose_kernel<<<dim3(SP / 32, HID / 32), dim3(32, 8), 0, stream>>>(W_dec, WdecT);

    encode_wmma_kernel<<<dim3(SP / 128, NROWS / 128), 256, 0, stream>>>(
        hB, WencB, b_enc, codes);

    const size_t topk_smem = ((size_t)SP + 256 + 64 + 64 + 8) * sizeof(unsigned int);
    topk_kernel<<<NROWS, 256, topk_smem, stream>>>(codes, idxf, kptr);

    if (use_tr)
        decode_t_kernel<<<NROWS, 256, 0, stream>>>(codes, idxf, WdecT, rec, kptr);
    else
        decode_nt_kernel<<<NROWS, 256, 0, stream>>>(codes, idxf, W_dec, rec, kptr);
}
